// EncoderDecoder_85710367359405
// MI455X (gfx1250) — compile-verified, matrix operand order checked by reading, not run on hardware
//
#include <hip/hip_runtime.h>
#include <math.h>

typedef __bf16 bf16;
typedef __attribute__((ext_vector_type(16))) __bf16 v16bf;
typedef __attribute__((ext_vector_type(8)))  float v8f;

#define Bq 32
#define Tq 400
#define Lq 20
#define Eq 256
#define Hq 512
#define Aq 64
#define Vq 50000
#define G3H 1536

union FragBF { v16bf v; unsigned u[8]; };

// Dual-A WMMA k-loop: acc0 += A0*W^T, acc1 += A1*W^T sharing the W fragment.
// Fragment layout per CDNA5 ISA 7.12.2 (16-bit A/B): lane r=lane&15 indexes the
// row (A: M, B: N); VGPR slot + lane-half select k; consecutive k-pairs allow
// one dword load per slot (compiler merges these into global_load_b128).
__device__ __forceinline__ void wmma_kloop2(const bf16* A0, const bf16* A1, int lda,
                                            const bf16* Wb, int ldw, int K, int lane,
                                            v8f& acc0, v8f& acc1) {
  const int r = lane & 15;
  const int hsel = (lane & 16) ? 8 : 0;
  for (int k0 = 0; k0 < K; k0 += 32) {
    FragBF fa0, fa1, fb;
#pragma unroll
    for (int v = 0; v < 8; ++v) {
      const int kk = k0 + ((v & 4) << 2) + hsel + ((v & 3) << 1);
      fb.u[v]  = *(const unsigned*)(Wb + (size_t)r * ldw + kk);
      fa0.u[v] = *(const unsigned*)(A0 + (size_t)r * lda + kk);
      fa1.u[v] = *(const unsigned*)(A1 + (size_t)r * lda + kk);
    }
    acc0 = __builtin_amdgcn_wmma_f32_16x16x32_bf16(false, fa0.v, false, fb.v,
                                                   (short)0, acc0, false, false);
    acc1 = __builtin_amdgcn_wmma_f32_16x16x32_bf16(false, fa1.v, false, fb.v,
                                                   (short)0, acc1, false, false);
  }
}

// ------- generic GEMM: C[M,N] = A[M,K] * W[N,K]^T, 32x16 output per wave ----
// M must be a multiple of 32, N of 16, K of 32. OUT dtype via template.
template <bool BF16OUT>
__global__ __launch_bounds__(256) void k_gemm_bf16(
    const bf16* __restrict__ A, int lda, const bf16* __restrict__ W, int ldw,
    void* __restrict__ C, long long ldc, int M, int N, int K) {
  const int lane = threadIdx.x & 31;
  const int wave = (blockIdx.x << 3) + (threadIdx.x >> 5);
  const int tn_cnt = N >> 4;
  if (wave >= (M >> 5) * tn_cnt) return;          // wave-uniform guard
  const int tm = wave / tn_cnt, tn = wave - tm * tn_cnt;
  v8f acc0 = {}, acc1 = {};
  const bf16* A0 = A + (size_t)(tm << 5) * lda;
  wmma_kloop2(A0, A0 + (size_t)16 * lda, lda,
              W + (size_t)(tn << 4) * ldw, ldw, K, lane, acc0, acc1);
  const int rbase = (lane & 16) ? 8 : 0;
  const int col = lane & 15;
  if constexpr (BF16OUT) {
    bf16* Cb = (bf16*)C + (size_t)(tm << 5) * ldc + (tn << 4);
#pragma unroll
    for (int v = 0; v < 8; ++v) {
      Cb[(size_t)(rbase + v) * ldc + col]      = (bf16)acc0[v];
      Cb[(size_t)(16 + rbase + v) * ldc + col] = (bf16)acc1[v];
    }
  } else {
    float* Cb = (float*)C + (size_t)(tm << 5) * ldc + (tn << 4);
#pragma unroll
    for (int v = 0; v < 8; ++v) {
      Cb[(size_t)(rbase + v) * ldc + col]      = acc0[v];
      Cb[(size_t)(16 + rbase + v) * ldc + col] = acc1[v];
    }
  }
}

// ---------------- helpers ---------------------------------------------------
__global__ __launch_bounds__(256) void k_f32_to_bf16(bf16* dst, const float* src, int n) {
  const int i = blockIdx.x * 256 + threadIdx.x;
  if (i < n) dst[i] = (bf16)src[i];
}

__global__ __launch_bounds__(256) void k_embed_src(
    const int* __restrict__ src, const float* __restrict__ embed,
    bf16* __restrict__ xs) {                 // xs[T][B][E]
  const int i = blockIdx.x * 256 + threadIdx.x;
  if (i >= Tq * Bq * Eq) return;
  const int e = i & (Eq - 1);
  const int tb = i >> 8;
  const int b = tb & 31, t = tb >> 5;
  const int tok = src[b * Tq + t];
  xs[i] = (bf16)embed[(size_t)tok * Eq + e];
}

// ------- encoder GRU scan (1 block/direction, 32 waves, gi precomputed) -----
// Serial critical path per step: gh = h*Whh^T (96 n-tile jobs, W fragment
// shared across both M tiles -> Whh read exactly once = 1.57MB from L2),
// then fused gate math with h held in registers.
__global__ __launch_bounds__(1024) void k_gru_scan(
    const bf16* __restrict__ gi_pre,      // [2][T][B][3H] precomputed x*Wih^T
    const bf16* __restrict__ whh_f, const bf16* __restrict__ whh_b,
    const float* __restrict__ bih_f, const float* __restrict__ bhh_f,
    const float* __restrict__ bih_b, const float* __restrict__ bhh_b,
    bf16* __restrict__ enc_hid,           // [B][T][2H]
    float* __restrict__ enc_final) {      // [B][2H]
  extern __shared__ __align__(16) char smem_raw[];
  bf16* sh_h  = (bf16*)smem_raw;          // B*H    (32KB)
  bf16* sh_gh = sh_h + Bq * Hq;           // B*3H   (96KB)
  const int dir = blockIdx.x;
  const bf16*  Whh = dir ? whh_b : whh_f;
  const float* bih = dir ? bih_b : bih_f;
  const float* bhh = dir ? bhh_b : bhh_f;
  const bf16*  gi_d = gi_pre + (size_t)dir * Tq * Bq * G3H;
  const int tid = threadIdx.x, lane = tid & 31, wave = tid >> 5;
  float hreg[16];
#pragma unroll
  for (int k = 0; k < 16; ++k) { hreg[k] = 0.0f; sh_h[tid * 16 + k] = (bf16)0.0f; }
  __syncthreads();
  for (int t = 0; t < Tq; ++t) {
    const int ti = dir ? (Tq - 1 - t) : t;
    // phase 1: gh = h * Whh^T  (96 jobs x 2 WMMA tiles, 3 jobs per wave)
    for (int tn = wave; tn < 96; tn += 32) {
      v8f acc0 = {}, acc1 = {};
      wmma_kloop2(sh_h, sh_h + 16 * Hq, Hq,
                  Whh + (size_t)(tn * 16) * Hq, Hq, Hq, lane, acc0, acc1);
      const int rbase = (lane & 16) ? 8 : 0, col = tn * 16 + (lane & 15);
#pragma unroll
      for (int v = 0; v < 8; ++v) {
        sh_gh[(rbase + v) * G3H + col]      = (bf16)acc0[v];
        sh_gh[(16 + rbase + v) * G3H + col] = (bf16)acc1[v];
      }
    }
    __syncthreads();
    // phase 2: fused GRU gates; gi slice streamed from the precomputed buffer
    const bf16* gi_t = gi_d + (size_t)ti * Bq * G3H;
#pragma unroll
    for (int k = 0; k < 16; ++k) {
      const int i = tid * 16 + k;
      const int b = i >> 9, j = i & 511;
      const float ir  = (float)gi_t[b * G3H + j]           + bih[j];
      const float iz  = (float)gi_t[b * G3H + Hq + j]      + bih[Hq + j];
      const float in_ = (float)gi_t[b * G3H + 2 * Hq + j]  + bih[2 * Hq + j];
      const float hr  = (float)sh_gh[b * G3H + j]          + bhh[j];
      const float hz  = (float)sh_gh[b * G3H + Hq + j]     + bhh[Hq + j];
      const float hn  = (float)sh_gh[b * G3H + 2 * Hq + j] + bhh[2 * Hq + j];
      const float r = 1.0f / (1.0f + expf(-(ir + hr)));
      const float z = 1.0f / (1.0f + expf(-(iz + hz)));
      const float n = tanhf(in_ + r * hn);
      const float h2 = (1.0f - z) * n + z * hreg[k];
      hreg[k] = h2;
      sh_h[i] = (bf16)h2;
      enc_hid[((size_t)b * Tq + ti) * (2 * Hq) + dir * Hq + j] = (bf16)h2;
    }
    __syncthreads();
  }
#pragma unroll
  for (int k = 0; k < 16; ++k) {
    const int i = tid * 16 + k;
    const int b = i >> 9, j = i & 511;
    enc_final[b * (2 * Hq) + dir * Hq + j] = hreg[k];
  }
}

// ---------------- attribute path (one block per batch row) ------------------
__global__ __launch_bounds__(256) void k_attr_path(
    const int* user, const int* product, const int* rating,
    const float* user_emb, const float* prod_emb, const float* rate_emb,
    const float* enc_final,
    const float* tfW, const float* tfb, const float* afW, const float* afb,
    const float* egW, const float* egb, const float* hwW, const float* hwb,
    const float* atta_key,
    float* attrs_out, float* highway_out, float* pk_attr,
    float* dec_h_f, bf16* dec_h_b, float* ch_f) {
  __shared__ float ef[2 * Hq], at[3 * Aq], tf[Hq], af[Hq], red[256];
  __shared__ float gsh;
  const int b = blockIdx.x, tid = threadIdx.x;
  for (int i = tid; i < 2 * Hq; i += 256) ef[i] = enc_final[b * 2 * Hq + i];
  for (int i = tid; i < 3 * Aq; i += 256) {
    float v;
    if (i < Aq)          v = user_emb[(size_t)user[b] * Aq + i];
    else if (i < 2 * Aq) v = prod_emb[(size_t)product[b] * Aq + (i - Aq)];
    else                 v = rate_emb[(size_t)rating[b] * Aq + (i - 2 * Aq)];
    at[i] = v; attrs_out[b * 3 * Aq + i] = v;
  }
  __syncthreads();
  for (int i = tid; i < Hq; i += 256) {
    float s1 = tfb[i], s2 = afb[i], s3 = hwb[i];
    for (int j = 0; j < 2 * Hq; ++j) s1 += tfW[i * 2 * Hq + j] * ef[j];
    for (int j = 0; j < 3 * Aq; ++j) {
      s2 += afW[i * 3 * Aq + j] * at[j];
      s3 += hwW[i * 3 * Aq + j] * at[j];
    }
    tf[i] = s1; af[i] = s2; highway_out[b * Hq + i] = s3;
  }
  float p = 0.f;
  for (int i = tid; i < 2 * Hq + 3 * Aq; i += 256)
    p += egW[i] * (i < 2 * Hq ? ef[i] : at[i - 2 * Hq]);
  red[tid] = p; __syncthreads();
  for (int s = 128; s > 0; s >>= 1) { if (tid < s) red[tid] += red[tid + s]; __syncthreads(); }
  if (tid == 0) gsh = 1.0f / (1.0f + expf(-(red[0] + egb[0])));
  __syncthreads();
  const float g = gsh;
  for (int i = tid; i < Hq; i += 256) {
    const float h = g * tf[i] + (1.f - g) * af[i];
    dec_h_f[b * Hq + i] = h; dec_h_b[b * Hq + i] = (bf16)h; ch_f[b * Hq + i] = h;
  }
  for (int i2 = tid; i2 < 3 * Hq; i2 += 256) {
    const int kk = i2 / Hq, i = i2 - kk * Hq;
    float s = 0.f;
    for (int j = 0; j < Aq; ++j) s += atta_key[i * Aq + j] * at[kk * Aq + j];
    pk_attr[(b * 3 + kk) * Hq + i] = s;
  }
}

// ---------------- decoder step kernels --------------------------------------
__global__ __launch_bounds__(256) void k_build_rnn_in(
    const int* trg, const float* embed, const float* ch, const float* highway,
    bf16* rnn_in, int l) {
  const int i = blockIdx.x * 256 + threadIdx.x;   // B*(E+2H)
  if (i >= Bq * (Eq + 2 * Hq)) return;
  const int b = i / (Eq + 2 * Hq), j = i - b * (Eq + 2 * Hq);
  float v;
  if (j < Eq) {
    const int tok = (l == 0) ? 1 : trg[b * Lq + l - 1];
    v = embed[(size_t)tok * Eq + j];
  } else if (j < Eq + Hq) v = ch[b * Hq + (j - Eq)];
  else                    v = highway[b * Hq + (j - Eq - Hq)];
  rnn_in[i] = (bf16)v;
}

__global__ __launch_bounds__(256) void k_gate_dec(
    const float* gi, const float* gh, const float* bih, const float* bhh,
    float* h_f, bf16* h_b) {
  const int i = blockIdx.x * 256 + threadIdx.x;   // B*H
  if (i >= Bq * Hq) return;
  const int b = i >> 9, j = i & 511;
  const float ir  = gi[b * G3H + j]          + bih[j];
  const float iz  = gi[b * G3H + Hq + j]     + bih[Hq + j];
  const float in_ = gi[b * G3H + 2 * Hq + j] + bih[2 * Hq + j];
  const float hr  = gh[b * G3H + j]          + bhh[j];
  const float hz  = gh[b * G3H + Hq + j]     + bhh[Hq + j];
  const float hn  = gh[b * G3H + 2 * Hq + j] + bhh[2 * Hq + j];
  const float r = 1.0f / (1.0f + expf(-(ir + hr)));
  const float z = 1.0f / (1.0f + expf(-(iz + hz)));
  const float n = tanhf(in_ + r * hn);
  const float h2 = (1.0f - z) * n + z * h_f[i];
  h_f[i] = h2; h_b[i] = (bf16)h2;
}

__global__ __launch_bounds__(256) void k_attn_text(
    const float* qq, const bf16* proj_key, const float* energy,
    const bf16* enc_hid, const int* src,
    float* ctx_text, float* al_all, int l) {
  __shared__ float qs[Hq], es[Hq], sc[Tq], red[256];
  __shared__ float s0, s1;
  const int b = blockIdx.x, tid = threadIdx.x;
  const int lane = tid & 31, wave = tid >> 5;
  for (int i = tid; i < Hq; i += 256) { qs[i] = qq[b * Hq + i]; es[i] = energy[i]; }
  __syncthreads();
  for (int t = wave; t < Tq; t += 8) {
    const bf16* pk = proj_key + ((size_t)b * Tq + t) * Hq;
    float p = 0.f;
    for (int j = lane; j < Hq; j += 32) p += tanhf(qs[j] + (float)pk[j]) * es[j];
    for (int m = 16; m > 0; m >>= 1) p += __shfl_xor(p, m, 32);
    if (lane == 0) sc[t] = (src[b * Tq + t] > 0) ? p : -INFINITY;
  }
  __syncthreads();
  float mv = -INFINITY;
  for (int t = tid; t < Tq; t += 256) mv = fmaxf(mv, sc[t]);
  red[tid] = mv; __syncthreads();
  for (int s = 128; s > 0; s >>= 1) { if (tid < s) red[tid] = fmaxf(red[tid], red[tid + s]); __syncthreads(); }
  if (tid == 0) s0 = red[0];
  __syncthreads();
  const float mx = s0;
  float sv = 0.f;
  for (int t = tid; t < Tq; t += 256) sv += expf(sc[t] - mx);
  red[tid] = sv; __syncthreads();
  for (int s = 128; s > 0; s >>= 1) { if (tid < s) red[tid] += red[tid + s]; __syncthreads(); }
  if (tid == 0) s1 = red[0];
  __syncthreads();
  const float inv = 1.0f / s1;
  for (int t = tid; t < Tq; t += 256) {
    const float a = expf(sc[t] - mx) * inv;
    sc[t] = a;
    al_all[((size_t)b * Lq + l) * Tq + t] = a;
  }
  __syncthreads();
  for (int d = tid; d < 2 * Hq; d += 256) {
    float s = 0.f;
    const bf16* eh = enc_hid + ((size_t)b * Tq) * (2 * Hq) + d;
    for (int t = 0; t < Tq; ++t) s += sc[t] * (float)eh[(size_t)t * (2 * Hq)];
    ctx_text[b * 2 * Hq + d] = s;
  }
}

__global__ __launch_bounds__(256) void k_dec_post(
    const float* dec_h, const float* ctx_text, const float* attrs, const float* pk_attr,
    const int* trg, const float* embed,
    const float* atta_query, const float* atta_energy,
    const float* tcW, const float* tcb, const float* acW, const float* acb,
    const float* cgW, const float* cgb, const float* chW, const float* chb,
    const float* gpW, const float* gpb,
    float* ch_state, bf16* ch2_all, float* gp_all, int l) {
  __shared__ float q[Hq], ct[2 * Hq], pv[Eq], qa[Hq], ca[Aq], cx[Hq], tcs[Hq], acs[Hq], red[256];
  __shared__ float sc3[3], al3[3], sclr;
  const int b = blockIdx.x, tid = threadIdx.x;
  const int lane = tid & 31, wave = tid >> 5;
  const int tok = (l == 0) ? 1 : trg[b * Lq + l - 1];
  for (int i = tid; i < Hq; i += 256)     q[i]  = dec_h[b * Hq + i];
  for (int i = tid; i < 2 * Hq; i += 256) ct[i] = ctx_text[b * 2 * Hq + i];
  for (int i = tid; i < Eq; i += 256)     pv[i] = embed[(size_t)tok * Eq + i];
  __syncthreads();
  for (int i = tid; i < Hq; i += 256) {
    float s = 0.f;
    for (int j = 0; j < Hq; ++j) s += atta_query[i * Hq + j] * q[j];
    qa[i] = s;
  }
  __syncthreads();
  if (wave < 3) {
    float p = 0.f;
    const float* pk = pk_attr + (b * 3 + wave) * Hq;
    for (int j = lane; j < Hq; j += 32) p += atta_energy[j] * tanhf(qa[j] + pk[j]);
    for (int m = 16; m > 0; m >>= 1) p += __shfl_xor(p, m, 32);
    if (lane == 0) sc3[wave] = p;
  }
  __syncthreads();
  if (tid == 0) {
    const float m3 = fmaxf(sc3[0], fmaxf(sc3[1], sc3[2]));
    const float e0 = expf(sc3[0] - m3), e1 = expf(sc3[1] - m3), e2 = expf(sc3[2] - m3);
    const float is = 1.f / (e0 + e1 + e2);
    al3[0] = e0 * is; al3[1] = e1 * is; al3[2] = e2 * is;
  }
  __syncthreads();
  if (tid < Aq)
    ca[tid] = al3[0] * attrs[b * 3 * Aq + tid] + al3[1] * attrs[b * 3 * Aq + Aq + tid]
            + al3[2] * attrs[b * 3 * Aq + 2 * Aq + tid];
  __syncthreads();
  for (int i = tid; i < Hq; i += 256) {
    float s1 = tcb[i];
    for (int j = 0; j < 2 * Hq; ++j) s1 += tcW[i * 2 * Hq + j] * ct[j];
    tcs[i] = s1;
    float s2 = acb[i];
    for (int j = 0; j < Aq; ++j) s2 += acW[i * Aq + j] * ca[j];
    acs[i] = s2;
  }
  float p = 0.f;  // cg over [q, prev, ctx_attr, ctx_text] = 1856
  for (int i = tid; i < 1856; i += 256) {
    float x;
    if (i < Hq)                x = q[i];
    else if (i < Hq + Eq)      x = pv[i - Hq];
    else if (i < Hq + Eq + Aq) x = ca[i - Hq - Eq];
    else                       x = ct[i - Hq - Eq - Aq];
    p += cgW[i] * x;
  }
  red[tid] = p; __syncthreads();
  for (int s = 128; s > 0; s >>= 1) { if (tid < s) red[tid] += red[tid + s]; __syncthreads(); }
  if (tid == 0) sclr = 1.f / (1.f + expf(-(red[0] + cgb[0])));
  __syncthreads();
  const float cg = sclr;
  for (int i = tid; i < Hq; i += 256) cx[i] = cg * tcs[i] + (1.f - cg) * acs[i];
  __syncthreads();
  for (int i = tid; i < Hq; i += 256) {
    float s = chb[i];
    for (int j = 0; j < Hq; ++j) s += chW[i * 2 * Hq + j] * q[j];
    for (int j = 0; j < Hq; ++j) s += chW[i * 2 * Hq + Hq + j] * cx[j];
    s = tanhf(s);
    ch_state[b * Hq + i] = s;
    ch2_all[((size_t)b * Lq + l) * Hq + i] = (bf16)s;
  }
  float p2 = 0.f;  // gp over [context, q, prev] = 1280
  for (int i = tid; i < 1280; i += 256) {
    const float x = (i < Hq) ? cx[i] : (i < 2 * Hq ? q[i - Hq] : pv[i - 2 * Hq]);
    p2 += gpW[i] * x;
  }
  red[tid] = p2; __syncthreads();
  for (int s = 128; s > 0; s >>= 1) { if (tid < s) red[tid] += red[tid + s]; __syncthreads(); }
  if (tid == 0) gp_all[b * Lq + l] = 1.f / (1.f + expf(-(red[0] + gpb[0])));
}

// ---------------- final softmax + pointer-generator mix ---------------------
__global__ __launch_bounds__(256) void k_softmax_mix(
    float* out, const float* gp_all, const float* al_all, const int* src) {
  __shared__ float red[256]; __shared__ float s0;
  const int row = blockIdx.x;            // b*Lq + l
  const int b = row / Lq;
  float* o = out + (size_t)row * Vq;
  const int tid = threadIdx.x;
  float mv = -INFINITY;
  for (int v = tid; v < Vq; v += 256) mv = fmaxf(mv, o[v]);
  red[tid] = mv; __syncthreads();
  for (int s = 128; s > 0; s >>= 1) { if (tid < s) red[tid] = fmaxf(red[tid], red[tid + s]); __syncthreads(); }
  if (tid == 0) s0 = red[0];
  __syncthreads();
  const float mx = s0;
  float sv = 0.f;
  for (int v = tid; v < Vq; v += 256) sv += expf(o[v] - mx);
  red[tid] = sv; __syncthreads();
  for (int s = 128; s > 0; s >>= 1) { if (tid < s) red[tid] += red[tid + s]; __syncthreads(); }
  if (tid == 0) s0 = red[0];
  __syncthreads();
  const float gp = gp_all[row];
  const float scale = gp / s0;
  for (int v = tid; v < Vq; v += 256) o[v] = expf(o[v] - mx) * scale;
  __syncthreads();
  const float cp = 1.0f - gp;
  const float* al = al_all + (size_t)row * Tq;
  for (int t = tid; t < Tq; t += 256)
    atomicAdd(&o[src[b * Tq + t]], cp * al[t]);
}

// ---------------- host orchestration ----------------------------------------
extern "C" void kernel_launch(void* const* d_in, const int* in_sizes, int n_in,
                              void* d_out, int out_size, void* d_ws, size_t ws_size,
                              hipStream_t stream) {
  enum {
    I_SRC = 0, I_SRC_, I_TRG, I_USER, I_PROD, I_RATE, I_VSZ,
    I_EMBED, I_EFWIH, I_EFWHH, I_EFBIH, I_EFBHH,
    I_EBWIH, I_EBWHH, I_EBBIH, I_EBBHH,
    I_TFW, I_TFB, I_UEMB, I_PEMB, I_REMB,
    I_AFW, I_AFB, I_EGW, I_EGB, I_HWW, I_HWB,
    I_DWIH, I_DWHH, I_DBIH, I_DBHH,
    I_ATTK, I_ATTQ, I_ATTE, I_ATAK, I_ATAQ, I_ATAE,
    I_TCW, I_TCB, I_ACW, I_ACB, I_CGW, I_CGB,
    I_CHW, I_CHB, I_GPW, I_GPB, I_GENW
  };
  auto F = [&](int i) { return (const float*)d_in[i]; };
  auto IP = [&](int i) { return (const int*)d_in[i]; };

  char* base = (char*)d_ws; size_t off = 0;
  auto alloc = [&](size_t bytes) -> void* {
    void* p = base + off; off += (bytes + 255) & ~(size_t)255; return p;
  };
  // activations / states
  bf16*  xs       = (bf16*)alloc((size_t)Tq * Bq * Eq * 2);
  bf16*  gi_pre   = (bf16*)alloc((size_t)2 * Tq * Bq * G3H * 2);
  bf16*  enc_hid  = (bf16*)alloc((size_t)Bq * Tq * 2 * Hq * 2);
  float* enc_fin  = (float*)alloc((size_t)Bq * 2 * Hq * 4);
  bf16*  proj_key = (bf16*)alloc((size_t)Bq * Tq * Hq * 2);
  float* attrs    = (float*)alloc((size_t)Bq * 3 * Aq * 4);
  float* highway  = (float*)alloc((size_t)Bq * Hq * 4);
  float* pk_attr  = (float*)alloc((size_t)Bq * 3 * Hq * 4);
  float* dec_h_f  = (float*)alloc((size_t)Bq * Hq * 4);
  bf16*  dec_h_b  = (bf16*)alloc((size_t)Bq * Hq * 2);
  float* ch_f     = (float*)alloc((size_t)Bq * Hq * 4);
  bf16*  rnn_in   = (bf16*)alloc((size_t)Bq * (Eq + 2 * Hq) * 2);
  float* gi_d     = (float*)alloc((size_t)Bq * G3H * 4);
  float* gh_d     = (float*)alloc((size_t)Bq * G3H * 4);
  float* qq       = (float*)alloc((size_t)Bq * Hq * 4);
  float* ctx_text = (float*)alloc((size_t)Bq * 2 * Hq * 4);
  bf16*  ch2_all  = (bf16*)alloc((size_t)Bq * Lq * Hq * 2);
  float* gp_all   = (float*)alloc((size_t)Bq * Lq * 4);
  float* al_all   = (float*)alloc((size_t)Bq * Lq * Tq * 4);
  // bf16 weight copies
  bf16* w_efwih = (bf16*)alloc((size_t)G3H * Eq * 2);
  bf16* w_efwhh = (bf16*)alloc((size_t)G3H * Hq * 2);
  bf16* w_ebwih = (bf16*)alloc((size_t)G3H * Eq * 2);
  bf16* w_ebwhh = (bf16*)alloc((size_t)G3H * Hq * 2);
  bf16* w_attk  = (bf16*)alloc((size_t)Hq * 2 * Hq * 2);
  bf16* w_attq  = (bf16*)alloc((size_t)Hq * Hq * 2);
  bf16* w_dwih  = (bf16*)alloc((size_t)G3H * (Eq + 2 * Hq) * 2);
  bf16* w_dwhh  = (bf16*)alloc((size_t)G3H * Hq * 2);
  bf16* w_gen   = (bf16*)alloc((size_t)Vq * Hq * 2);

  auto conv = [&](bf16* dst, const float* src, int n) {
    k_f32_to_bf16<<<(n + 255) / 256, 256, 0, stream>>>(dst, src, n);
  };
  conv(w_efwih, F(I_EFWIH), G3H * Eq);
  conv(w_efwhh, F(I_EFWHH), G3H * Hq);
  conv(w_ebwih, F(I_EBWIH), G3H * Eq);
  conv(w_ebwhh, F(I_EBWHH), G3H * Hq);
  conv(w_attk,  F(I_ATTK),  Hq * 2 * Hq);
  conv(w_attq,  F(I_ATTQ),  Hq * Hq);
  conv(w_dwih,  F(I_DWIH),  G3H * (Eq + 2 * Hq));
  conv(w_dwhh,  F(I_DWHH),  G3H * Hq);
  conv(w_gen,   F(I_GENW),  Vq * Hq);

  k_embed_src<<<(Tq * Bq * Eq + 255) / 256, 256, 0, stream>>>(IP(I_SRC_), F(I_EMBED), xs);

  auto gemmF = [&](const bf16* A, int lda, const bf16* W, int ldw,
                   float* C, long long ldc, int M, int N, int K) {
    const int jobs = (M >> 5) * (N >> 4);
    k_gemm_bf16<false><<<(jobs + 7) / 8, 256, 0, stream>>>(A, lda, W, ldw, C, ldc, M, N, K);
  };
  auto gemmB = [&](const bf16* A, int lda, const bf16* W, int ldw,
                   bf16* C, long long ldc, int M, int N, int K) {
    const int jobs = (M >> 5) * (N >> 4);
    k_gemm_bf16<true><<<(jobs + 7) / 8, 256, 0, stream>>>(A, lda, W, ldw, C, ldc, M, N, K);
  };

  // gi = xs @ Wih^T for all timesteps, both directions (parallel, off the
  // serial critical path)
  gemmB(xs, Eq, w_efwih, Eq, gi_pre,                          G3H, Tq * Bq, G3H, Eq);
  gemmB(xs, Eq, w_ebwih, Eq, gi_pre + (size_t)Tq * Bq * G3H,  G3H, Tq * Bq, G3H, Eq);

  const size_t scan_lds = (size_t)(Bq * Hq + Bq * G3H) * sizeof(bf16); // 128KB
  k_gru_scan<<<2, 1024, scan_lds, stream>>>(
      gi_pre, w_efwhh, w_ebwhh,
      F(I_EFBIH), F(I_EFBHH), F(I_EBBIH), F(I_EBBHH), enc_hid, enc_fin);

  k_attr_path<<<Bq, 256, 0, stream>>>(
      IP(I_USER), IP(I_PROD), IP(I_RATE),
      F(I_UEMB), F(I_PEMB), F(I_REMB), enc_fin,
      F(I_TFW), F(I_TFB), F(I_AFW), F(I_AFB),
      F(I_EGW), F(I_EGB), F(I_HWW), F(I_HWB), F(I_ATAK),
      attrs, highway, pk_attr, dec_h_f, dec_h_b, ch_f);

  // proj_key = enc_hidden @ att_key^T (bf16 out: re-read 20x during decode)
  gemmB(enc_hid, 2 * Hq, w_attk, 2 * Hq, proj_key, Hq, Bq * Tq, Hq, 2 * Hq);

  for (int l = 0; l < Lq; ++l) {
    k_build_rnn_in<<<(Bq * (Eq + 2 * Hq) + 255) / 256, 256, 0, stream>>>(
        IP(I_TRG), F(I_EMBED), ch_f, highway, rnn_in, l);
    gemmF(rnn_in, Eq + 2 * Hq, w_dwih, Eq + 2 * Hq, gi_d, G3H, Bq, G3H, Eq + 2 * Hq);
    gemmF(dec_h_b, Hq, w_dwhh, Hq, gh_d, G3H, Bq, G3H, Hq);
    k_gate_dec<<<(Bq * Hq + 255) / 256, 256, 0, stream>>>(
        gi_d, gh_d, F(I_DBIH), F(I_DBHH), dec_h_f, dec_h_b);
    gemmF(dec_h_b, Hq, w_attq, Hq, qq, Hq, Bq, Hq, Hq);
    k_attn_text<<<Bq, 256, 0, stream>>>(
        qq, proj_key, F(I_ATTE), enc_hid, IP(I_SRC), ctx_text, al_all, l);
    k_dec_post<<<Bq, 256, 0, stream>>>(
        dec_h_f, ctx_text, attrs, pk_attr, IP(I_TRG), F(I_EMBED),
        F(I_ATAQ), F(I_ATAE), F(I_TCW), F(I_TCB), F(I_ACW), F(I_ACB),
        F(I_CGW), F(I_CGB), F(I_CHW), F(I_CHB), F(I_GPW), F(I_GPB),
        ch_f, ch2_all, gp_all, l);
  }

  // generator logits straight into d_out [B*L, V], then softmax+copy mix
  gemmF(ch2_all, Hq, w_gen, Hq, (float*)d_out, Vq, Bq * Lq, Vq, Hq);
  k_softmax_mix<<<Bq * Lq, 256, 0, stream>>>((float*)d_out, gp_all, al_all, IP(I_SRC));
}
